// RNN_79568564126524
// MI455X (gfx1250) — matrix operand order repaired and checked by reading, not compiled, not run on hardware
//
#include <hip/hip_runtime.h>
#include <hip/hip_bf16.h>
#include <math.h>

#define VOCAB  32000
#define HIDDEN 512
#define BATCH  32
#define SEQ    256
#define RWG    16          // workgroups cooperating on the recurrence

typedef __attribute__((ext_vector_type(16))) _Float16 v16h;
typedef __attribute__((ext_vector_type(8)))  float    v8f;
typedef __attribute__((ext_vector_type(4)))  float    v4f;

union Frag {
    v16h h;
    v4f  f4[2];
    _Float16 e[16];
};

// branch-free tanh: ~6 VALU (v_exp_f32 + v_rcp_f32), error ~1 ulp
__device__ __forceinline__ float tanh_fast(float v) {
    float ax = fabsf(v);
    float e  = __expf(ax + ax);                       // exp(2|x|); inf -> th=1
    float th = 1.0f - 2.0f * __builtin_amdgcn_rcpf(e + 1.0f);
    return copysignf(th, v);
}

// ---------------------------------------------------------------------------
// Split fp32 weight W[K][N] into f16 hi/lo, stored transposed: T[n][k].
// ---------------------------------------------------------------------------
__global__ void split_transpose_kernel(const float* __restrict__ W,
                                       _Float16* __restrict__ Thi,
                                       _Float16* __restrict__ Tlo,
                                       int K, int N) {
    long long idx   = (long long)blockIdx.x * blockDim.x + threadIdx.x;
    long long total = (long long)K * N;
    if (idx >= total) return;
    int n = (int)(idx % N);
    int k = (int)(idx / N);
    float v = W[(long long)k * N + n];
    _Float16 hi = (_Float16)v;
    _Float16 lo = (_Float16)(v - (float)hi);
    Thi[(long long)n * K + k] = hi;
    Tlo[(long long)n * K + k] = lo;
}

// Row-major f16 hi/lo split (used to seed Aext rows 0..31 with H0).
__global__ void split_rows_kernel(const float* __restrict__ src,
                                  _Float16* __restrict__ hi,
                                  _Float16* __restrict__ lo,
                                  long long total) {
    long long i = (long long)blockIdx.x * blockDim.x + threadIdx.x;
    if (i >= total) return;
    float v = src[i];
    _Float16 h = (_Float16)v;
    hi[i] = h;
    lo[i] = (_Float16)(v - (float)h);
}

// Fully parallel embedding gather (+ hidden bias), hoisted out of the
// sequential recurrence: Emb[t*B+b][h] = W_xh[x[b][t]][h] + b_h[h].
__global__ void emb_kernel(const int* __restrict__ x,
                           const float* __restrict__ W_xh,
                           const float* __restrict__ b_h,
                           float* __restrict__ Emb) {
    long long i = (long long)blockIdx.x * blockDim.x + threadIdx.x;
    if (i >= (long long)SEQ * BATCH * HIDDEN) return;
    int h = (int)(i % HIDDEN);
    long long row = i / HIDDEN;          // t*B + b
    int b = (int)(row & (BATCH - 1));
    int t = (int)(row >> 5);
    int tok = x[b * SEQ + t];
    Emb[i] = W_xh[(long long)tok * HIDDEN + h] + b_h[h];
}

__global__ void zero_u32_kernel(unsigned* __restrict__ p) {
    if (threadIdx.x == 0 && blockIdx.x == 0) *p = 0u;
}

// ---------------------------------------------------------------------------
// Monotonic inter-workgroup barrier (persistent kernel, RWG blocks).
// ---------------------------------------------------------------------------
__device__ __forceinline__ void grid_barrier(unsigned* ctr, unsigned target) {
    __threadfence();           // release this WG's stores
    __syncthreads();
    if (threadIdx.x == 0) {
        atomicAdd(ctr, 1u);
        while (__hip_atomic_load(ctr, __ATOMIC_RELAXED,
                                 __HIP_MEMORY_SCOPE_AGENT) < target) {}
    }
    __syncthreads();
    __threadfence();           // acquire: don't serve stale cached rows
}

// ---------------------------------------------------------------------------
// Recurrence across 16 persistent workgroups. WG w owns hidden columns
// [w*32, w*32+32); its W_hh hi/lo slice stays resident in 64 KB LDS for all
// 256 steps. The recurrent state IS the f16 hi/lo history (Aext): step t
// reads rows t*32.. (previous H, split) and writes rows (t+1)*32.. ; rows
// 0..31 hold split H0. One spin barrier per step; emb values prefetched
// before the K loop so their latency hides under the WMMAs.
// ---------------------------------------------------------------------------
__global__ __launch_bounds__(128)
void rnn_recurrence_kernel(const float* __restrict__ Emb,
                           const _Float16* __restrict__ whh_t_hi,
                           const _Float16* __restrict__ whh_t_lo,
                           _Float16* __restrict__ Aext_hi,
                           _Float16* __restrict__ Aext_lo,
                           float* __restrict__ H_final_out,
                           unsigned* __restrict__ barrier_ctr) {
    __shared__ _Float16 Bhi[32][HIDDEN];   // 32 KB
    __shared__ _Float16 Blo[32][HIDDEN];   // 32 KB

    const int wg    = blockIdx.x;          // 0..15
    const int tid   = threadIdx.x;
    const int lane  = tid & 31;
    const int wave  = tid >> 5;            // 0..3
    const int mt    = wave & 1;
    const int nt    = wave >> 1;           // 0..1
    const int lm    = lane & 15;
    const int lk    = lane >> 4;
    const int ncol0 = wg * 32;

    // one-time fill of the LDS-resident W_hh slice
    {
        const v4f* sh = (const v4f*)(whh_t_hi + (size_t)ncol0 * HIDDEN);
        const v4f* sl = (const v4f*)(whh_t_lo + (size_t)ncol0 * HIDDEN);
        v4f* dh = (v4f*)&Bhi[0][0];
        v4f* dl = (v4f*)&Blo[0][0];
        for (int i = tid; i < 32 * HIDDEN / 8; i += 128) {
            dh[i] = sh[i];
            dl[i] = sl[i];
        }
    }
    __syncthreads();

    const int m = mt * 16 + lm;            // batch row this lane reads for A
    const int n = ncol0 + nt * 16 + lm;    // global hidden column of C tile
    const int brow = mt * 16 + lk * 8;     // first batch row of this lane's C

    for (int t = 0; t < SEQ; ++t) {
        const size_t arow = (size_t)t * BATCH;

        // prefetch emb(+bias) values; latency overlaps the K loop below
        float ev[8];
        #pragma unroll
        for (int r = 0; r < 8; ++r)
            ev[r] = Emb[(arow + brow + r) * HIDDEN + n];

        v8f c = {};
        for (int kc = 0; kc < HIDDEN; kc += 32) {
            // A fragment: previous-step H already stored as f16 hi/lo
            const size_t ro = (arow + m) * HIDDEN + kc + lk * 8;
            Frag ah, al;
            ah.f4[0] = *(const v4f*)(Aext_hi + ro);
            ah.f4[1] = *(const v4f*)(Aext_hi + ro + 16);
            al.f4[0] = *(const v4f*)(Aext_lo + ro);
            al.f4[1] = *(const v4f*)(Aext_lo + ro + 16);
            // B fragment from LDS-resident weight slice
            const _Float16* bhp = &Bhi[nt * 16 + lm][kc + lk * 8];
            const _Float16* blp = &Blo[nt * 16 + lm][kc + lk * 8];
            Frag bh, bl;
            bh.f4[0] = *(const v4f*)(bhp);
            bh.f4[1] = *(const v4f*)(bhp + 16);
            bl.f4[0] = *(const v4f*)(blp);
            bl.f4[1] = *(const v4f*)(blp + 16);

            c = __builtin_amdgcn_wmma_f32_16x16x32_f16(false, ah.h, false, bh.h, (short)0, c, false, false);
            c = __builtin_amdgcn_wmma_f32_16x16x32_f16(false, al.h, false, bh.h, (short)0, c, false, false);
            c = __builtin_amdgcn_wmma_f32_16x16x32_f16(false, ah.h, false, bl.h, (short)0, c, false, false);
        }

        // epilogue: branch-free tanh, write split H for step t+1 / GEMM
        #pragma unroll
        for (int r = 0; r < 8; ++r) {
            int b   = brow + r;
            float h = tanh_fast(c[r] + ev[r]);
            _Float16 hi = (_Float16)h;
            size_t wrow = (arow + BATCH + b) * HIDDEN + n;
            Aext_hi[wrow] = hi;
            Aext_lo[wrow] = (_Float16)(h - (float)hi);
            if (t == SEQ - 1) H_final_out[(size_t)b * HIDDEN + n] = h;
        }

        grid_barrier(barrier_ctr, (unsigned)RWG * (unsigned)(t + 1));
    }
}

// ---------------------------------------------------------------------------
// Output GEMM: out[8192, 32000] = Hhist @ W_hy + b_y.
// 256 threads/block (8 waves), block tile 64x128, wave tile 32x32 (2x2 frags).
// blockIdx.x sweeps M so consecutive blocks reuse the same B panel in L2.
// ---------------------------------------------------------------------------
__global__ __launch_bounds__(256)
void output_gemm_kernel(const _Float16* __restrict__ A_hi,
                        const _Float16* __restrict__ A_lo,
                        const _Float16* __restrict__ B_hi,   // [VOCAB][HIDDEN]
                        const _Float16* __restrict__ B_lo,
                        const float* __restrict__ b_y,
                        float* __restrict__ out) {
    const int m0   = blockIdx.x * 64;
    const int n0   = blockIdx.y * 128;
    const int lane = threadIdx.x & 31;
    const int wave = threadIdx.x >> 5;
    const int wm   = (wave >> 2) * 32;
    const int wn   = (wave & 3) * 32;
    const int lm   = lane & 15;
    const int lk   = lane >> 4;

    v8f acc[2][2] = {};

    for (int kc = 0; kc < HIDDEN; kc += 32) {
        Frag ah[2], al[2], bh[2], bl[2];
        #pragma unroll
        for (int i = 0; i < 2; ++i) {
            long long row = (long long)(m0 + wm + i * 16 + lm) * HIDDEN + kc + lk * 8;
            const _Float16* ph = A_hi + row;
            const _Float16* pl = A_lo + row;
            ah[i].f4[0] = *(const v4f*)(ph);
            ah[i].f4[1] = *(const v4f*)(ph + 16);
            al[i].f4[0] = *(const v4f*)(pl);
            al[i].f4[1] = *(const v4f*)(pl + 16);
        }
        #pragma unroll
        for (int j = 0; j < 2; ++j) {
            long long row = (long long)(n0 + wn + j * 16 + lm) * HIDDEN + kc + lk * 8;
            const _Float16* ph = B_hi + row;
            const _Float16* pl = B_lo + row;
            bh[j].f4[0] = *(const v4f*)(ph);
            bh[j].f4[1] = *(const v4f*)(ph + 16);
            bl[j].f4[0] = *(const v4f*)(pl);
            bl[j].f4[1] = *(const v4f*)(pl + 16);
        }
        #pragma unroll
        for (int i = 0; i < 2; ++i)
            #pragma unroll
            for (int j = 0; j < 2; ++j) {
                acc[i][j] = __builtin_amdgcn_wmma_f32_16x16x32_f16(false, ah[i].h, false, bh[j].h, (short)0, acc[i][j], false, false);
                acc[i][j] = __builtin_amdgcn_wmma_f32_16x16x32_f16(false, al[i].h, false, bh[j].h, (short)0, acc[i][j], false, false);
                acc[i][j] = __builtin_amdgcn_wmma_f32_16x16x32_f16(false, ah[i].h, false, bl[j].h, (short)0, acc[i][j], false, false);
            }
    }

    #pragma unroll
    for (int i = 0; i < 2; ++i) {
        #pragma unroll
        for (int j = 0; j < 2; ++j) {
            int n = n0 + wn + j * 16 + lm;
            float bias = b_y[n];
            #pragma unroll
            for (int r = 0; r < 8; ++r) {
                int m = m0 + wm + i * 16 + lk * 8 + r;
                out[(long long)m * VOCAB + n] = acc[i][j][r] + bias;
            }
        }
    }
}

// ---------------------------------------------------------------------------
extern "C" void kernel_launch(void* const* d_in, const int* in_sizes, int n_in,
                              void* d_out, int out_size, void* d_ws, size_t ws_size,
                              hipStream_t stream) {
    const int*   x    = (const int*)d_in[0];
    const float* H0   = (const float*)d_in[1];
    const float* W_xh = (const float*)d_in[2];
    const float* W_hh = (const float*)d_in[3];
    const float* b_h  = (const float*)d_in[4];
    const float* W_hy = (const float*)d_in[5];
    const float* b_y  = (const float*)d_in[6];

    float* out     = (float*)d_out;
    float* H_final = out + (long long)SEQ * BATCH * VOCAB;   // tuple tail

    const size_t AROWS = (size_t)(SEQ + 1) * BATCH;          // +32 rows for H0

    // workspace layout
    char* ws = (char*)d_ws;
    _Float16* whh_t_hi = (_Float16*)ws;  ws += (size_t)HIDDEN * HIDDEN * 2;
    _Float16* whh_t_lo = (_Float16*)ws;  ws += (size_t)HIDDEN * HIDDEN * 2;
    _Float16* why_t_hi = (_Float16*)ws;  ws += (size_t)VOCAB  * HIDDEN * 2;
    _Float16* why_t_lo = (_Float16*)ws;  ws += (size_t)VOCAB  * HIDDEN * 2;
    _Float16* Aext_hi  = (_Float16*)ws;  ws += AROWS * HIDDEN * 2;
    _Float16* Aext_lo  = (_Float16*)ws;  ws += AROWS * HIDDEN * 2;
    float*    Emb      = (float*)ws;     ws += (size_t)SEQ * BATCH * HIDDEN * 4;
    unsigned* bar_ctr  = (unsigned*)ws;  ws += 256;

    // 0) reset the inter-WG barrier counter (deterministic per launch)
    zero_u32_kernel<<<1, 1, 0, stream>>>(bar_ctr);

    // 1) split+transpose recurrent and output weights into f16 hi/lo
    {
        long long tot = (long long)HIDDEN * HIDDEN;
        split_transpose_kernel<<<(int)((tot + 255) / 256), 256, 0, stream>>>(
            W_hh, whh_t_hi, whh_t_lo, HIDDEN, HIDDEN);
    }
    {
        long long tot = (long long)HIDDEN * VOCAB;
        split_transpose_kernel<<<(int)((tot + 255) / 256), 256, 0, stream>>>(
            W_hy, why_t_hi, why_t_lo, HIDDEN, VOCAB);
    }

    // 2) seed Aext rows 0..31 with split H0
    {
        long long tot = (long long)BATCH * HIDDEN;
        split_rows_kernel<<<(int)((tot + 255) / 256), 256, 0, stream>>>(
            H0, Aext_hi, Aext_lo, tot);
    }

    // 3) fully parallel embedding gather (+ b_h), off the critical path
    {
        long long tot = (long long)SEQ * BATCH * HIDDEN;
        emb_kernel<<<(int)((tot + 255) / 256), 256, 0, stream>>>(
            x, W_xh, b_h, Emb);
    }

    // 4) sequential recurrence: 16 persistent WGs, LDS-resident W_hh slices
    rnn_recurrence_kernel<<<RWG, 128, 0, stream>>>(
        Emb, whh_t_hi, whh_t_lo, Aext_hi, Aext_lo, H_final, bar_ctr);

    // 5) big output GEMM: [8192 x 512] @ [512 x 32000]
    dim3 grid(8192 / 64, VOCAB / 128);
    output_gemm_kernel<<<grid, 256, 0, stream>>>(
        Aext_hi + (size_t)BATCH * HIDDEN, Aext_lo + (size_t)BATCH * HIDDEN,
        why_t_hi, why_t_lo, b_y, out);
}
